// MHSA_9337258901572
// MI455X (gfx1250) — matrix operand order repaired and checked by reading, not compiled, not run on hardware
//
#include <hip/hip_runtime.h>
#include <hip/hip_bf16.h>

typedef __attribute__((ext_vector_type(16))) _Float16 v16h;
typedef __attribute__((ext_vector_type(8)))  _Float16 v8h;
typedef __attribute__((ext_vector_type(8)))  float    v8f;

#define B_   2
#define S_   2048
#define H_   16
#define G_   4
#define DH_  64
#define D_   1024
#define DKV_ 256
#define SCALE_ 0.125f  // 1/sqrt(64)

#define WMMA_F16(a, b, c) \
  __builtin_amdgcn_wmma_f32_16x16x32_f16(false, (a), false, (b), (short)0, (c), false, false)

// A-fragment (16x32 f16, row = lane%16): elems 0..15 = A[m][k + 16*kh .. +15]
// (contiguous k order, matching the B loads)
__device__ __forceinline__ v16h loadA_frag(const _Float16* row, int kbase, int kh) {
  v8h lo = *(const v8h*)(row + kbase + kh * 16);
  v8h hi = *(const v8h*)(row + kbase + kh * 16 + 8);
  return __builtin_shufflevector(lo, hi, 0, 1, 2, 3, 4, 5, 6, 7, 8, 9, 10, 11, 12, 13, 14, 15);
}

// ---------------------------------------------------------------------------
// f32 -> f16 convert
// ---------------------------------------------------------------------------
__global__ void cvt_f32_f16(const float* __restrict__ in, _Float16* __restrict__ out, int n) {
  int i = blockIdx.x * blockDim.x + threadIdx.x;
  int stride = gridDim.x * blockDim.x;
  for (; i < n; i += stride) out[i] = (_Float16)in[i];
}

// ---------------------------------------------------------------------------
// WMMA GEMM: C[m,n] = sum_k A[m,k] * Bw[n,k]   (A:[4096,1024], Bw:[N,1024])
// One wave -> 32x64 tile (2 M-subtiles x 4 N-subtiles => 8 WMMA per K=32 step,
// 12 b128 loads per step: B fragments reused across both M-subtiles).
// mode: 0=Q(+rope), 1=K(+rope), 2=V(transposed store), 3=O(f32 out)
// ---------------------------------------------------------------------------
__global__ __launch_bounds__(256) void gemm_qkvo(
    const _Float16* __restrict__ A, const _Float16* __restrict__ Bw,
    const float* __restrict__ rope, int N, int mode,
    _Float16* __restrict__ outH, float* __restrict__ outF) {
  int lane = threadIdx.x & 31;
  int m16 = lane & 15, kh = lane >> 4;
  int wid = (blockIdx.x * blockDim.x + threadIdx.x) >> 5;
  int tilesN = N >> 6;
  int mt = wid / tilesN, nt = wid % tilesN;
  int nb = nt * 64;
  const _Float16* Arow0 = A + (size_t)(mt * 32 + m16) * D_;
  const _Float16* Arow1 = A + (size_t)(mt * 32 + 16 + m16) * D_;

  v8f c00 = {}, c01 = {}, c02 = {}, c03 = {};
  v8f c10 = {}, c11 = {}, c12 = {}, c13 = {};
  for (int k = 0; k < D_; k += 32) {
    v16h a0 = loadA_frag(Arow0, k, kh);
    v16h a1 = loadA_frag(Arow1, k, kh);
    v16h b0 = *(const v16h*)(Bw + (size_t)(nb + 0 + m16) * D_ + k + kh * 16);
    v16h b1 = *(const v16h*)(Bw + (size_t)(nb + 16 + m16) * D_ + k + kh * 16);
    v16h b2 = *(const v16h*)(Bw + (size_t)(nb + 32 + m16) * D_ + k + kh * 16);
    v16h b3 = *(const v16h*)(Bw + (size_t)(nb + 48 + m16) * D_ + k + kh * 16);
    c00 = WMMA_F16(a0, b0, c00);
    c01 = WMMA_F16(a0, b1, c01);
    c02 = WMMA_F16(a0, b2, c02);
    c03 = WMMA_F16(a0, b3, c03);
    c10 = WMMA_F16(a1, b0, c10);
    c11 = WMMA_F16(a1, b1, c11);
    c12 = WMMA_F16(a1, b2, c12);
    c13 = WMMA_F16(a1, b3, c13);
  }

#pragma unroll
  for (int i = 0; i < 2; ++i) {
#pragma unroll
    for (int j = 0; j < 4; ++j) {
      v8f c = (i == 0) ? ((j == 0) ? c00 : (j == 1) ? c01 : (j == 2) ? c02 : c03)
                       : ((j == 0) ? c10 : (j == 1) ? c11 : (j == 2) ? c12 : c13);
      int n = nb + j * 16 + m16;
#pragma unroll
      for (int r = 0; r < 8; ++r) {
        float v = c[r];
        int m = mt * 32 + i * 16 + r + 8 * kh;  // global row = b*S + s
        int s = m & (S_ - 1);
        int b = m >> 11;
        if (mode <= 1) {  // RoPE: partner column n^1 sits in lane^1 of this fragment
          int d = n & (DH_ - 1);
          float cs = rope[s * DH_ + d];
          float sn = rope[S_ * DH_ + s * DH_ + d];
          float other = __shfl_xor(v, 1, 32);
          v = v * cs + ((d & 1) ? other : -other) * sn;
        }
        if (mode == 0) {        // Q: [B,H,S,DH] f16
          int h = n >> 6, d = n & 63;
          outH[(((size_t)(b * H_ + h)) * S_ + s) * DH_ + d] = (_Float16)v;
        } else if (mode == 1) { // K: [B,G,S,DH] f16
          int g = n >> 6, d = n & 63;
          outH[(((size_t)(b * G_ + g)) * S_ + s) * DH_ + d] = (_Float16)v;
        } else if (mode == 2) { // V transposed: [B,G,DH,S] f16
          int g = n >> 6, d = n & 63;
          outH[(((size_t)(b * G_ + g)) * DH_ + d) * S_ + s] = (_Float16)v;
        } else {                // O: f32 row-major [B*S, D]
          outF[(size_t)m * D_ + n] = v;
        }
      }
    }
  }
}

// ---------------------------------------------------------------------------
// Flash attention: one wave per 16-query tile, online softmax over 32-key chunks.
// ---------------------------------------------------------------------------
__global__ __launch_bounds__(256) void attn_fa(
    const _Float16* __restrict__ Qh,  // [B,H,S,DH]
    const _Float16* __restrict__ Kh,  // [B,G,S,DH]
    const _Float16* __restrict__ Vt,  // [B,G,DH,S]
    _Float16* __restrict__ Yh) {      // [B*S, D]
  __shared__ _Float16 lds[8][16 * 32];  // per-wave P bounce buffer
  int lane = threadIdx.x & 31;
  int m16 = lane & 15, kh = lane >> 4;
  int wave = threadIdx.x >> 5;
  int wid = blockIdx.x * 8 + wave;        // 4096 waves total
  int qt = wid & (S_ / 16 - 1);
  int bh = wid >> 7;
  int h = bh & (H_ - 1), b = bh >> 4;
  int g = h >> 2;                          // repeat_interleave GQA mapping
  int qlo = qt << 4;
  const _Float16* Kb = Kh + ((size_t)(b * G_ + g)) * S_ * DH_;
  const _Float16* Vb = Vt + ((size_t)(b * G_ + g)) * DH_ * S_;
  _Float16* ldsw = &lds[wave][0];

  const _Float16* Qrow =
      Qh + ((size_t)(b * H_ + h)) * S_ * DH_ + (size_t)(qlo + m16) * DH_;
  v16h aQ0 = loadA_frag(Qrow, 0, kh);   // d 0..31
  v16h aQ1 = loadA_frag(Qrow, 32, kh);  // d 32..63

  float mi[8], li[8];
  v8f acc0 = {}, acc1 = {}, acc2 = {}, acc3 = {};
#pragma unroll
  for (int r = 0; r < 8; ++r) { mi[r] = -1e30f; li[r] = 0.0f; }

  for (int k0 = 0; k0 < qlo + 16; k0 += 32) {
    bool masked = (k0 + 32 > qlo);
    if (k0 + 32 < qlo + 16)  // pull next K chunk toward L2 (global_prefetch_b8)
      __builtin_prefetch(Kb + (size_t)(k0 + 32 + m16) * DH_, 0, 1);

    // S = Q K^T  (two 16-key subtiles, K-dim = DH = 64 -> 2 WMMA each)
    v8f s0 = {}, s1 = {};
    {
      v16h bk = *(const v16h*)(Kb + (size_t)(k0 + m16) * DH_ + kh * 16);
      s0 = WMMA_F16(aQ0, bk, s0);
      bk = *(const v16h*)(Kb + (size_t)(k0 + m16) * DH_ + 32 + kh * 16);
      s0 = WMMA_F16(aQ1, bk, s0);
      bk = *(const v16h*)(Kb + (size_t)(k0 + 16 + m16) * DH_ + kh * 16);
      s1 = WMMA_F16(aQ0, bk, s1);
      bk = *(const v16h*)(Kb + (size_t)(k0 + 16 + m16) * DH_ + 32 + kh * 16);
      s1 = WMMA_F16(aQ1, bk, s1);
    }

    // online softmax (row lives across the lane's 16-lane half; shfl_xor 1..8)
#pragma unroll
    for (int r = 0; r < 8; ++r) {
      float f0 = s0[r] * SCALE_, f1 = s1[r] * SCALE_;
      if (masked) {
        int q = qlo + r + 8 * kh;
        if (k0 + m16 > q) f0 = -__builtin_inff();
        if (k0 + 16 + m16 > q) f1 = -__builtin_inff();
      }
      float rm = fmaxf(f0, f1);
      rm = fmaxf(rm, __shfl_xor(rm, 1, 32));
      rm = fmaxf(rm, __shfl_xor(rm, 2, 32));
      rm = fmaxf(rm, __shfl_xor(rm, 4, 32));
      rm = fmaxf(rm, __shfl_xor(rm, 8, 32));
      float mn = fmaxf(mi[r], rm);
      float corr = __expf(mi[r] - mn);
      float e0 = __expf(f0 - mn), e1 = __expf(f1 - mn);
      float rs = e0 + e1;
      rs += __shfl_xor(rs, 1, 32);
      rs += __shfl_xor(rs, 2, 32);
      rs += __shfl_xor(rs, 4, 32);
      rs += __shfl_xor(rs, 8, 32);
      li[r] = li[r] * corr + rs;
      mi[r] = mn;
      acc0[r] *= corr; acc1[r] *= corr; acc2[r] *= corr; acc3[r] *= corr;
      int ql = r + 8 * kh;  // C-layout -> A-layout transpose via LDS
      ldsw[ql * 32 + m16] = (_Float16)e0;
      ldsw[ql * 32 + 16 + m16] = (_Float16)e1;
    }

    // P (16 x 32 keys) as A-fragment from LDS, then acc += P * V
    v16h aP = loadA_frag(ldsw + m16 * 32, 0, kh);
    v16h bv;
    bv = *(const v16h*)(Vb + (size_t)(0 * 16 + m16) * S_ + k0 + kh * 16);
    acc0 = WMMA_F16(aP, bv, acc0);
    bv = *(const v16h*)(Vb + (size_t)(1 * 16 + m16) * S_ + k0 + kh * 16);
    acc1 = WMMA_F16(aP, bv, acc1);
    bv = *(const v16h*)(Vb + (size_t)(2 * 16 + m16) * S_ + k0 + kh * 16);
    acc2 = WMMA_F16(aP, bv, acc2);
    bv = *(const v16h*)(Vb + (size_t)(3 * 16 + m16) * S_ + k0 + kh * 16);
    acc3 = WMMA_F16(aP, bv, acc3);
  }

#pragma unroll
  for (int r = 0; r < 8; ++r) {
    float inv = 1.0f / li[r];
    int srow = qlo + r + 8 * kh;
    _Float16* yrow = Yh + ((size_t)b * S_ + srow) * D_ + h * DH_;
    yrow[0 * 16 + m16] = (_Float16)(acc0[r] * inv);
    yrow[1 * 16 + m16] = (_Float16)(acc1[r] * inv);
    yrow[2 * 16 + m16] = (_Float16)(acc2[r] * inv);
    yrow[3 * 16 + m16] = (_Float16)(acc3[r] * inv);
  }
}

// ---------------------------------------------------------------------------
extern "C" void kernel_launch(void* const* d_in, const int* in_sizes, int n_in,
                              void* d_out, int out_size, void* d_ws, size_t ws_size,
                              hipStream_t stream) {
  (void)in_sizes; (void)n_in; (void)out_size; (void)ws_size;
  const float* X    = (const float*)d_in[0];
  const float* Wq   = (const float*)d_in[1];
  const float* Wk   = (const float*)d_in[2];
  const float* Wv   = (const float*)d_in[3];
  const float* Wo   = (const float*)d_in[4];
  const float* rope = (const float*)d_in[5];
  // d_in[6] (causal mask) unused: causality applied analytically.

  _Float16* p = (_Float16*)d_ws;
  _Float16* Xh  = p; p += (size_t)4096 * 1024;          // X       f16
  _Float16* Wqh = p; p += (size_t)1024 * 1024;          // Wq      f16
  _Float16* Wkh = p; p += (size_t)256 * 1024;           // Wk      f16
  _Float16* Wvh = p; p += (size_t)256 * 1024;           // Wv      f16
  _Float16* Woh = p; p += (size_t)1024 * 1024;          // Wo      f16
  _Float16* Qh  = p; p += (size_t)B_ * H_ * S_ * DH_;   // Q roped [B,H,S,DH]
  _Float16* Kh  = p; p += (size_t)B_ * G_ * S_ * DH_;   // K roped [B,G,S,DH]
  _Float16* Vt  = p; p += (size_t)B_ * G_ * S_ * DH_;   // V^T     [B,G,DH,S]
  _Float16* Yh  = p;                                    // attn out [B*S, D]

  cvt_f32_f16<<<512, 256, 0, stream>>>(X, Xh, 4096 * 1024);
  cvt_f32_f16<<<256, 256, 0, stream>>>(Wq, Wqh, 1024 * 1024);
  cvt_f32_f16<<<128, 256, 0, stream>>>(Wk, Wkh, 256 * 1024);
  cvt_f32_f16<<<128, 256, 0, stream>>>(Wv, Wvh, 256 * 1024);
  cvt_f32_f16<<<256, 256, 0, stream>>>(Wo, Woh, 1024 * 1024);

  // waves = (4096/32) * (N/64); 8 waves per block of 256 threads
  gemm_qkvo<<<256, 256, 0, stream>>>(Xh, Wqh, rope, 1024, 0, Qh, nullptr);
  gemm_qkvo<<<64, 256, 0, stream>>>(Xh, Wkh, rope, 256, 1, Kh, nullptr);
  gemm_qkvo<<<64, 256, 0, stream>>>(Xh, Wvh, nullptr, 256, 2, Vt, nullptr);

  attn_fa<<<512, 256, 0, stream>>>(Qh, Kh, Vt, Yh);

  gemm_qkvo<<<256, 256, 0, stream>>>(Yh, Woh, nullptr, 1024, 3, nullptr, (float*)d_out);
}
